// GATLayer_4930622456156
// MI455X (gfx1250) — matrix operand order repaired and checked by reading, not compiled, not run on hardware
//
#include <hip/hip_runtime.h>
#include <hip/hip_bf16.h>

#define NEG_SLOPE 0.2f

typedef __attribute__((ext_vector_type(16))) __bf16 v16bf;
typedef __attribute__((ext_vector_type(8)))  float  v8f;

static constexpr int Bb = 8, Nn = 2048, Dd = 64;

// leaky(x) = max(x, 0.2x)  (valid since 0 < slope < 1)
__device__ __forceinline__ float leaky(float x) { return fmaxf(x, NEG_SLOPE * x); }

// ---------------------------------------------------------------------------
// K1: Wh = h @ W^T  (fp32 math), fused s_i = Wh·a1, s_j = Wh·a2,
//     and pack Wh to bf16 in WMMA-B-fragment layout (this file's contiguous k order, same as A):
//     element (b, j, col):  jc=j>>5, k=j&31, c=col>>4, n=col&15,
//     lane = n + ((k>>4)<<4), e = k&15
//     pidx = (((b*64 + jc)*4 + c)*32 + lane)*16 + e
// ---------------------------------------------------------------------------
__global__ __launch_bounds__(256)
void k1_wh(const float* __restrict__ h, const float* __restrict__ W,
           const float* __restrict__ a,
           __bf16* __restrict__ whp, float* __restrict__ si, float* __restrict__ sj)
{
    __shared__ float Wt[64 * 65];        // W transposed, padded (conflict-free)
    __shared__ float hr[4 * 64];
    __shared__ float red[2][4][64];
    const int t = threadIdx.x;
    const int row0 = blockIdx.x * 4;     // global row base (b*N + n)

    for (int idx = t; idx < 4096; idx += 256) {
        int o = idx >> 6, d = idx & 63;
        Wt[d * 65 + o] = W[idx];         // coalesced read, padded scatter write
    }
    {
        int r = t >> 6, d = t & 63;
        hr[r * 64 + d] = h[(size_t)(row0 + r) * 64 + d];
    }
    __syncthreads();

    const int r = t >> 6, o = t & 63;
    float acc = 0.f;
#pragma unroll
    for (int d = 0; d < 64; ++d)
        acc = fmaf(hr[r * 64 + d], Wt[d * 65 + o], acc);   // broadcast + conflict-free

    const int grow = row0 + r;           // = b*N + n
    {   // packed bf16 store (B-fragment layout), n plays the K/j role, o the N/col role
        int b  = grow >> 11;
        int n  = grow & 2047;
        int jc = n >> 5, k = n & 31;
        int c  = o >> 4, nn = o & 15;
        int lane = nn + ((k >> 4) << 4);
        int e  = k & 15;
        size_t pidx = ((size_t)((b * 64 + jc) * 4 + c) * 32 + lane) * 16 + e;
        whp[pidx] = (__bf16)acc;
    }
    red[0][r][o] = acc * a[o];
    red[1][r][o] = acc * a[64 + o];
    __syncthreads();
    if (t < 8) {
        int rr = t >> 1, which = t & 1;
        float s = 0.f;
#pragma unroll 4
        for (int q = 0; q < 64; ++q) s += red[which][rr][q];
        if (which == 0) si[row0 + rr] = s;
        else            sj[row0 + rr] = s;
    }
}

// ---------------------------------------------------------------------------
// K3: per-batch max of s_j (monotonic leaky => exact softmax row max is free)
// ---------------------------------------------------------------------------
__global__ __launch_bounds__(256)
void k3_max(const float* __restrict__ sj, float* __restrict__ mx)
{
    __shared__ float red[256];
    const int b = blockIdx.x, t = threadIdx.x;
    float m = -3.4e38f;
    for (int j = t; j < Nn; j += 256) m = fmaxf(m, sj[b * Nn + j]);
    red[t] = m;
    __syncthreads();
    for (int s = 128; s > 0; s >>= 1) {
        if (t < s) red[t] = fmaxf(red[t], red[t + s]);
        __syncthreads();
    }
    if (t == 0) mx[b] = red[0];
}

// ---------------------------------------------------------------------------
// K4: per-row c = m + log(sum_j exp(leaky(si+sj) - m)),  m = leaky(si + max_j sj)
//     so later alpha_ij = exp(leaky(si+sj) - c)  (one fused constant per row)
// ---------------------------------------------------------------------------
__global__ __launch_bounds__(256)
void k4_denom(const float* __restrict__ si, const float* __restrict__ sj,
              const float* __restrict__ mx, float* __restrict__ crow)
{
    __shared__ float sjl[Nn];            // 8 KB, broadcast reads
    const int t = threadIdx.x;
    const int b = blockIdx.x >> 3;       // 8 blocks per batch
    const int i = ((blockIdx.x & 7) << 8) + t;
    for (int j = t; j < Nn; j += 256) sjl[j] = sj[b * Nn + j];
    __syncthreads();

    const float s = si[b * Nn + i];
    const float m = leaky(s + mx[b]);
    float d0 = 0.f, d1 = 0.f, d2 = 0.f, d3 = 0.f;
    for (int j = 0; j < Nn; j += 4) {
        d0 += __expf(leaky(s + sjl[j + 0]) - m);
        d1 += __expf(leaky(s + sjl[j + 1]) - m);
        d2 += __expf(leaky(s + sjl[j + 2]) - m);
        d3 += __expf(leaky(s + sjl[j + 3]) - m);
    }
    crow[b * Nn + i] = m + __logf((d0 + d1) + (d2 + d3));
}

// ---------------------------------------------------------------------------
// K5: hot loop. One wave per 16-row i-tile (1024 waves, 4 waves/block).
// Per 32-wide j-chunk: coalesced fp32 alpha store + bf16 LDS staging,
// reload as WMMA A-fragment, 4x v_wmma_f32_16x16x32_bf16 vs packed Wh.
// ---------------------------------------------------------------------------
__global__ __launch_bounds__(128)
void k5_attn(const float* __restrict__ si, const float* __restrict__ sj,
             const float* __restrict__ crow, const __bf16* __restrict__ whp,
             float* __restrict__ hp, float* __restrict__ alpha)
{
    // per-wave 16x32 bf16 tile, row stride 40 halves (80B: 16B-aligned, conflict-free b128 reads)
    __shared__ __bf16 tile[4][16 * 40];

    const int t = threadIdx.x;
    const int lane = t & 31, wid = t >> 5;
    const int tileId = blockIdx.x * 4 + wid;   // 0..1023
    const int b  = tileId >> 7;                // 128 tiles per batch
    const int i0 = (tileId & 127) << 4;
    const int bn0 = b * Nn + i0;
    const int half = lane >> 4, rsel = lane & 15;

    // broadcast per-row constants into SGPR arrays via readlane
    float si_v = 0.f, c_v = 0.f;
    if (lane < 16) { si_v = si[bn0 + lane]; c_v = crow[bn0 + lane]; }
    float sir[16], cr[16];
#pragma unroll
    for (int r = 0; r < 16; ++r) {
        sir[r] = __int_as_float(__builtin_amdgcn_readlane(__float_as_int(si_v), r));
        cr[r]  = __int_as_float(__builtin_amdgcn_readlane(__float_as_int(c_v),  r));
    }

    const v8f vzero = {0.f, 0.f, 0.f, 0.f, 0.f, 0.f, 0.f, 0.f};
    v8f acc[4];
#pragma unroll
    for (int c = 0; c < 4; ++c) acc[c] = vzero;

    const float* sjb  = sj + b * Nn;
    float*       arow = alpha + (size_t)bn0 * Nn;
    __bf16*      myt  = tile[wid];

    for (int jc = 0; jc < 64; ++jc) {
        const int jbase = jc << 5;

        // ---- P1: alpha for 16 rows x 32 cols; lane owns one column (coalesced) ----
        const float sjv = sjb[jbase + lane];
#pragma unroll
        for (int r = 0; r < 16; ++r) {
            float av = __expf(leaky(sir[r] + sjv) - cr[r]);   // normalized alpha
            arow[(size_t)r * Nn + jbase + lane] = av;          // 128B/row coalesced
            myt[r * 40 + lane] = (__bf16)av;                   // stage for WMMA A
        }
        __syncthreads();

        // ---- P2: WMMA. A fragment: row=rsel, K = {half*16..+15} ----
        union Frag { v16bf v; uint4 q[2]; } au, bu;
        au.q[0] = *(const uint4*)(myt + rsel * 40 +     half * 16);
        au.q[1] = *(const uint4*)(myt + rsel * 40 + 8 + half * 16);

        const __bf16* bbase = whp + ((size_t)((b * 64 + jc) * 128) + lane) * 16;
        if (jc < 63) __builtin_prefetch(bbase + 2048, 0, 1);   // next chunk's B data

#pragma unroll
        for (int c = 0; c < 4; ++c) {
            const uint4* bp = (const uint4*)(bbase + (size_t)c * 512);
            bu.q[0] = bp[0];
            bu.q[1] = bp[1];
            acc[c] = __builtin_amdgcn_wmma_f32_16x16x32_bf16(
                false, au.v, false, bu.v, (short)0, acc[c], false, false);
        }
        __syncthreads();   // protect LDS tile before next P1 overwrites it
    }

    // ---- epilogue: C/D layout -> h_prime. VGPR r: M = r + half*8, N = rsel ----
#pragma unroll
    for (int c = 0; c < 4; ++c) {
#pragma unroll
        for (int r = 0; r < 8; ++r) {
            int row = i0 + r + half * 8;
            hp[((size_t)(b * Nn + row)) * 64 + c * 16 + rsel] = acc[c][r];
        }
    }
}

// ---------------------------------------------------------------------------
extern "C" void kernel_launch(void* const* d_in, const int* in_sizes, int n_in,
                              void* d_out, int out_size, void* d_ws, size_t ws_size,
                              hipStream_t stream)
{
    const float* h = (const float*)d_in[0];   // (8,2048,64)
    const float* W = (const float*)d_in[1];   // (64,64)
    const float* a = (const float*)d_in[2];   // (128,)

    char* ws = (char*)d_ws;
    __bf16* whp  = (__bf16*)(ws);                              // 2 MB packed Wh
    float*  si   = (float*)(ws + (2u << 20));                  // 64 KB
    float*  sj   = si + Bb * Nn;                               // 64 KB
    float*  crow = sj + Bb * Nn;                               // 64 KB
    float*  mx   = crow + Bb * Nn;                             // 32 B

    float* hp    = (float*)d_out;                              // (8,2048,64)
    float* alpha = hp + (size_t)Bb * Nn * Dd;                  // (8,2048,2048)

    k1_wh   <<<Bb * Nn / 4, 256, 0, stream>>>(h, W, a, whp, si, sj);
    k3_max  <<<Bb,          256, 0, stream>>>(sj, mx);
    k4_denom<<<Bb * 8,      256, 0, stream>>>(si, sj, mx, crow);
    k5_attn <<<256,         128, 0, stream>>>(si, sj, crow, whp, hp, alpha);
}
